// TempFlowNetwork_80625126081080
// MI455X (gfx1250) — matrix operand order repaired and checked by reading, not compiled, not run on hardware
//
#include <hip/hip_runtime.h>
#include <math.h>

// ---------------------------------------------------------------------------
// CDNA5 (gfx1250) implementation of TempFlowNetwork reference.
// All matmuls via V_WMMA_F32_16X16X32_F16 (wave32, f32 accumulate).
// Weights and activations are staged as _Float16 in workspace (K padded to
// multiples of 32), so the GEMM inner loop is pure global_load_b128 -> v_wmma
// with no conversions and no guards.
// ---------------------------------------------------------------------------

typedef __attribute__((ext_vector_type(16))) _Float16 v16h;
typedef __attribute__((ext_vector_type(8)))  float    v8f;

namespace cfg {
constexpr int B    = 256;
constexpr int D    = 742;
constexpr int Dp   = 768;      // D padded to mult of 32
constexpr int T    = 128;
constexpr int H    = 256;      // HDIM (already mult of 32)
constexpr int HID  = 100;
constexpr int HIDp = 128;      // HID padded
constexpr int NB   = 5;
constexpr int NS   = 100;
constexpr int SB   = NS * B;   // 25600
constexpr int WGT  = 256;      // threads per block (8 waves)
}

union F16x16 { v16h v; float4 f4[2]; };   // 16 halves = two 16B chunks

// ---------------------------------------------------------------------------
// Guard-free f16 WMMA GEMM:  C[M,N] = act( [A0|A1] @ W^T + Cin + bias )
//   A0: [M, K0] f16, K0 % 32 == 0 (row index optionally mod rowModA0)
//   A1: [M, Kt-K0] f16 (null if Kt==K0)
//   W : [N, Kt] f16, Kt % 32 == 0, zero-padded where logical K < Kt
//   C : f32 (outHalf=0) or f16 (outHalf=1)
//   Requirements: M % 16 == 0; all row strides multiples of 8 halves.
// Each wave computes a 16x64 output tile; two straight-line K loops.
// Fragment layouts follow CDNA5 ISA 7.12.2 (16-bit A 16x32, B 32x16, f32 C).
// ---------------------------------------------------------------------------
__global__ __launch_bounds__(cfg::WGT)
void k_gemm(const _Float16* __restrict__ A0, int lda0, int K0, int rowModA0,
            const _Float16* __restrict__ A1, int lda1, int Kt,
            const _Float16* __restrict__ W,  int ldw,
            const float* __restrict__ bias,
            const float* __restrict__ Cin, int ldcin,
            void* __restrict__ C, int ldc,
            int M, int N, int act, int outHalf)
{
  const int wave = (blockIdx.x * blockDim.x + threadIdx.x) >> 5;
  const int lane = threadIdx.x & 31;
  const int nt64 = (N + 63) >> 6;
  const int mt   = (M + 15) >> 4;
  const int tm   = wave / nt64;
  const int tn   = wave % nt64;
  if (tm >= mt) return;                 // wave-uniform exit (EXEC stays full)

  const int row0 = tm << 4;
  const int col0 = tn << 6;

  const int lrow = lane & 15;
  const int lhi  = lane >> 4;
  const int offA = lhi * 8;             // A-fragment K offset (halves)
  const int offB = lhi * 8;             // B-fragment K offset (halves)
  const int arow  = row0 + lrow;        // M % 16 == 0 -> always in range
  const int arow0 = rowModA0 ? (arow % rowModA0) : arow;

  // A row pointers, indexed directly by global k
  const _Float16* pA0 = A0 + (size_t)arow0 * lda0;
  const _Float16* pA1 = A1 ? (A1 + (size_t)arow * lda1 - K0) : pA0;

  // B row pointers (columns clamped for loads; stores are guarded)
  const int c0 = col0 + lrow;
  const _Float16* pB0 = W + (size_t)((c0      < N) ? c0      : N - 1) * ldw;
  const _Float16* pB1 = W + (size_t)((c0 + 16 < N) ? c0 + 16 : N - 1) * ldw;
  const _Float16* pB2 = W + (size_t)((c0 + 32 < N) ? c0 + 32 : N - 1) * ldw;
  const _Float16* pB3 = W + (size_t)((c0 + 48 < N) ? c0 + 48 : N - 1) * ldw;

  v8f acc0 = v8f{}, acc1 = v8f{}, acc2 = v8f{}, acc3 = v8f{};

  // one K-step: issue all 10 b128 loads, then 4 WMMAs.
  // A layout: elems 0-7 = K offA..offA+7, elems 8-15 = K offA+16..offA+23.
  // B layout: elems 0-7 = K offB..offB+7, elems 8-15 = K offB+16..offB+23.
  auto kstep = [&](const _Float16* pa, int k0) {
    const _Float16* pak = pa + k0;
    const _Float16* w0p = pB0 + k0 + offB;
    const _Float16* w1p = pB1 + k0 + offB;
    const _Float16* w2p = pB2 + k0 + offB;
    const _Float16* w3p = pB3 + k0 + offB;
    F16x16 a, b0, b1, b2, b3;
    a.f4[0]  = *reinterpret_cast<const float4*>(pak + offA);
    a.f4[1]  = *reinterpret_cast<const float4*>(pak + offA + 16);
    b0.f4[0] = *reinterpret_cast<const float4*>(w0p);
    b0.f4[1] = *reinterpret_cast<const float4*>(w0p + 16);
    b1.f4[0] = *reinterpret_cast<const float4*>(w1p);
    b1.f4[1] = *reinterpret_cast<const float4*>(w1p + 16);
    b2.f4[0] = *reinterpret_cast<const float4*>(w2p);
    b2.f4[1] = *reinterpret_cast<const float4*>(w2p + 16);
    b3.f4[0] = *reinterpret_cast<const float4*>(w3p);
    b3.f4[1] = *reinterpret_cast<const float4*>(w3p + 16);
    __builtin_prefetch(pak + offA + 32, 0, 1);
    __builtin_prefetch(w0p + 32, 0, 1);
    __builtin_prefetch(w1p + 32, 0, 1);
    __builtin_prefetch(w2p + 32, 0, 1);
    __builtin_prefetch(w3p + 32, 0, 1);
    acc0 = __builtin_amdgcn_wmma_f32_16x16x32_f16(false, a.v, false, b0.v, (short)0, acc0, false, false);
    acc1 = __builtin_amdgcn_wmma_f32_16x16x32_f16(false, a.v, false, b1.v, (short)0, acc1, false, false);
    acc2 = __builtin_amdgcn_wmma_f32_16x16x32_f16(false, a.v, false, b2.v, (short)0, acc2, false, false);
    acc3 = __builtin_amdgcn_wmma_f32_16x16x32_f16(false, a.v, false, b3.v, (short)0, acc3, false, false);
  };

  for (int k0 = 0;  k0 < K0; k0 += 32) kstep(pA0, k0);
  for (int k0 = K0; k0 < Kt; k0 += 32) kstep(pA1, k0);

  // ---- epilogue: VGPR j -> row (row0 + j + lhi*8), col (col0 + t*16 + lrow)
#pragma unroll
  for (int t = 0; t < 4; ++t) {
    const v8f acc = (t == 0) ? acc0 : (t == 1) ? acc1 : (t == 2) ? acc2 : acc3;
    const int col = col0 + t * 16 + lrow;
    if (col >= N) continue;
#pragma unroll
    for (int j = 0; j < 8; ++j) {
      const int row = row0 + j + lhi * 8;
      float v = acc[j];
      if (Cin)  v += Cin[(size_t)row * ldcin + col];
      if (bias) v += bias[col];
      if (act == 1)      v = tanhf(v);
      else if (act == 2) v = fmaxf(v, 0.0f);
      if (outHalf) ((_Float16*)C)[(size_t)row * ldc + col] = (_Float16)v;
      else         ((float*)C)[(size_t)row * ldc + col] = v;
    }
  }
}

// ---------------------------------------------------------------------------
// Elementwise / utility kernels
// ---------------------------------------------------------------------------
__global__ void k_fill(float* p, long long n, float v) {
  long long i = (long long)blockIdx.x * blockDim.x + threadIdx.x;
  if (i < n) p[i] = v;
}

__global__ void k_fillh(_Float16* p, long long n) {
  long long i = (long long)blockIdx.x * blockDim.x + threadIdx.x;
  if (i < n) p[i] = (_Float16)0.0f;
}

// zero-padded f32 -> f16 weight staging: Wp[n*Kp+k] = k<K ? W[n*K+k] : 0
__global__ void k_padW(const float* __restrict__ W, _Float16* __restrict__ Wp,
                       int N, int K, int Kp) {
  long long n = (long long)N * Kp;
  long long i = (long long)blockIdx.x * blockDim.x + threadIdx.x;
  if (i >= n) return;
  int k = (int)(i % Kp);
  int r = (int)(i / Kp);
  Wp[i] = (k < K) ? (_Float16)W[(size_t)r * K + k] : (_Float16)0.0f;
}

// src[r*D+d] -> dst[r*Dp+d] (f32), pad cols zeroed
__global__ void k_copy_pad(float* __restrict__ dst, const float* __restrict__ src,
                           long long rows, int Dn, int Dpn) {
  long long n = rows * Dpn;
  long long i = (long long)blockIdx.x * blockDim.x + threadIdx.x;
  if (i >= n) return;
  int d = (int)(i % Dpn);
  long long r = i / Dpn;
  dst[i] = (d < Dn) ? src[r * Dn + d] : 0.0f;
}

// past[B,D,T] -> xbt16[(b*T+t)*Dp + d], pad cols zeroed
__global__ void k_transpose_past(const float* __restrict__ past,
                                 _Float16* __restrict__ xbt,
                                 int Bn, int Dn, int Dpn, int Tn) {
  long long n = (long long)Bn * Tn * Dpn;
  long long i = (long long)blockIdx.x * blockDim.x + threadIdx.x;
  if (i >= n) return;
  int d = (int)(i % Dpn);
  long long r = i / Dpn;
  int t = (int)(r % Tn);
  int b = (int)(r / Tn);
  xbt[i] = (d < Dn) ? (_Float16)past[((size_t)b * Dn + d) * Tn + t] : (_Float16)0.0f;
}

// LSTM cell: g[B,4H] f32 -> c f32, h f16 ; optional f16 store into h_all
__global__ void k_lstm_cell(const float* __restrict__ g,
                            const float* __restrict__ cprev,
                            float* __restrict__ cout,
                            _Float16* __restrict__ hout,
                            _Float16* __restrict__ hstore, int strideB,
                            int n, int H4, int Hn) {
  int i = blockIdx.x * blockDim.x + threadIdx.x;
  if (i >= n) return;
  int b = i / Hn, j = i % Hn;
  const float* gr = g + (size_t)b * H4;
  float gi = gr[j], gf = gr[Hn + j], gg = gr[2 * Hn + j], go = gr[3 * Hn + j];
  float si = 1.0f / (1.0f + expf(-gi));
  float sf = 1.0f / (1.0f + expf(-gf));
  float so = 1.0f / (1.0f + expf(-go));
  float c  = sf * cprev[i] + si * tanhf(gg);
  float h  = so * tanhf(c);
  cout[i] = c;
  hout[i] = (_Float16)h;
  if (hstore) hstore[(size_t)b * strideB + j] = (_Float16)h;
}

// mx = u * blk_mask, f32 -> f16  (mask_d = ((d&1)==par) ? 1 : 0); pads zeroed
__global__ void k_maskmul(const float* __restrict__ u, _Float16* __restrict__ mx,
                          long long n, int Dn, int Dpn, int par) {
  long long i = (long long)blockIdx.x * blockDim.x + threadIdx.x;
  if (i >= n) return;
  int d = (int)(i % Dpn);
  mx[i] = (d < Dn && (d & 1) == par) ? (_Float16)u[i] : (_Float16)0.0f;
}

// forward coupling: u = mx + (1-m)*(u-t)*exp(-s);  sumld += -(1-m)*s
__global__ void k_cpl_fwd(float* __restrict__ u, float* __restrict__ sumld,
                          const float* __restrict__ s, const float* __restrict__ t,
                          long long n, int Dn, int Dpn, int par) {
  long long i = (long long)blockIdx.x * blockDim.x + threadIdx.x;
  if (i >= n) return;
  int d = (int)(i % Dpn);
  if (d < Dn && (d & 1) != par) {             // m == 0 half
    u[i] = (u[i] - t[i]) * expf(-s[i]);
    sumld[i] -= s[i];
  }
}

// inverse coupling: u = mu + (1-m)*(u*exp(s)+t)
__global__ void k_cpl_inv(float* __restrict__ u,
                          const float* __restrict__ s, const float* __restrict__ t,
                          long long n, int Dn, int Dpn, int par) {
  long long i = (long long)blockIdx.x * blockDim.x + threadIdx.x;
  if (i >= n) return;
  int d = (int)(i % Dpn);
  if (d < Dn && (d & 1) != par) u[i] = u[i] * expf(s[i]) + t[i];
}

__device__ inline unsigned hash_u32(unsigned x) {
  x ^= x >> 16; x *= 0x7feb352dU;
  x ^= x >> 15; x *= 0x846ca68bU;
  x ^= x >> 16; return x;
}

// deterministic standard normals (Box-Muller over hashed counter)
__global__ void k_randn(float* __restrict__ u, long long n, unsigned seed) {
  long long i = (long long)blockIdx.x * blockDim.x + threadIdx.x;
  if (i >= n) return;
  unsigned h1 = hash_u32((unsigned)(2 * i) ^ seed);
  unsigned h2 = hash_u32((unsigned)(2 * i + 1) ^ (seed * 0x9E3779B9u + 1u));
  float u1 = ((float)(h1 >> 8) + 1.0f) * (1.0f / 16777216.0f);  // (0,1]
  float u2 = (float)(h2 >> 8) * (1.0f / 16777216.0f);           // [0,1)
  float r  = sqrtf(-2.0f * logf(u1));
  u[i] = r * cosf(6.2831853071795864f * u2);
}

// loss accumulate: -(base + sumld)*mask / B over logical [B,D]
__global__ void k_loss_reduce(const float* __restrict__ u,
                              const float* __restrict__ sumld,
                              const float* __restrict__ mask,
                              float* __restrict__ out,
                              long long n, int Dn, int Dpn, float scale) {
  __shared__ float sm[cfg::WGT];
  float acc = 0.0f;
  for (long long i = (long long)blockIdx.x * blockDim.x + threadIdx.x;
       i < n; i += (long long)gridDim.x * blockDim.x) {
    long long r = i / Dn;
    int d = (int)(i % Dn);
    size_t ip = (size_t)r * Dpn + d;
    float uu = u[ip];
    float base = -0.5f * (uu * uu + 1.8378770664093453f);  // log(2*pi)
    acc += -(base + sumld[ip]) * mask[i];
  }
  sm[threadIdx.x] = acc;
  __syncthreads();
  for (int s = cfg::WGT / 2; s > 0; s >>= 1) {
    if (threadIdx.x < s) sm[threadIdx.x] += sm[threadIdx.x + s];
    __syncthreads();
  }
  if (threadIdx.x == 0) atomicAdd(out, sm[0] * scale);
}

// preds[s*B+b][d] = u[(s*B+b)*Dp + d] * mask[b][d]   (preds contiguous [.,D])
__global__ void k_store_preds(const float* __restrict__ u,
                              const float* __restrict__ mask,
                              float* __restrict__ preds,
                              long long n, int Dn, int Dpn, int Bn) {
  long long i = (long long)blockIdx.x * blockDim.x + threadIdx.x;
  if (i >= n) return;
  int d = (int)(i % Dn);
  long long r = i / Dn;
  int b = (int)(r % Bn);
  preds[i] = u[(size_t)r * Dpn + d] * mask[(size_t)b * Dn + d];
}

// mse accumulate: (mean_s preds - tgt)^2 * mask / (B*D)
__global__ void k_mse_reduce(const float* __restrict__ preds,
                             const float* __restrict__ tgt,
                             const float* __restrict__ mask,
                             float* __restrict__ out,
                             int BD, int S, float scale) {
  __shared__ float sm[cfg::WGT];
  float acc = 0.0f;
  for (int e = blockIdx.x * blockDim.x + threadIdx.x; e < BD;
       e += gridDim.x * blockDim.x) {
    float sum = 0.0f;
    for (int s = 0; s < S; ++s) sum += preds[(size_t)s * BD + e];
    float diff = sum / (float)S - tgt[e];
    acc += diff * diff * mask[e];
  }
  sm[threadIdx.x] = acc;
  __syncthreads();
  for (int s = cfg::WGT / 2; s > 0; s >>= 1) {
    if (threadIdx.x < s) sm[threadIdx.x] += sm[threadIdx.x + s];
    __syncthreads();
  }
  if (threadIdx.x == 0) atomicAdd(out, sm[0] * scale);
}

// ---------------------------------------------------------------------------
// Host orchestration
// ---------------------------------------------------------------------------
extern "C" void kernel_launch(void* const* d_in, const int* in_sizes, int n_in,
                              void* d_out, int out_size, void* d_ws, size_t ws_size,
                              hipStream_t stream) {
  using namespace cfg;
  (void)in_sizes; (void)n_in; (void)out_size; (void)ws_size;

  const float* past = (const float*)d_in[0];   // [B,D,T]
  const float* tgt  = (const float*)d_in[1];   // [B,D]
  const float* mask = (const float*)d_in[2];   // [B,D]
  const float* Wih0 = (const float*)d_in[3];   // [4H, D]
  const float* Whh0 = (const float*)d_in[4];   // [4H, H]
  const float* bih0 = (const float*)d_in[5];
  const float* bhh0 = (const float*)d_in[6];
  const float* Wih1 = (const float*)d_in[7];   // [4H, H]
  const float* Whh1 = (const float*)d_in[8];
  const float* bih1 = (const float*)d_in[9];
  const float* bhh1 = (const float*)d_in[10];
  auto fw = [&](int blk, int net, int l) -> const float* {
    return (const float*)d_in[11 + blk * 20 + net * 10 + l * 2];
  };
  auto fb = [&](int blk, int net, int l) -> const float* {
    return (const float*)d_in[11 + blk * 20 + net * 10 + l * 2 + 1];
  };

  float* out   = (float*)d_out;
  float* lossp = out;
  float* msep  = out + 1;
  float* preds = out + 2;                      // [NS, B, D]

  auto G1 = [](long long n) { return dim3((unsigned)((n + WGT - 1) / WGT)); };

  // -------- workspace layout (bump allocator in float units, 32B aligned) ---
  float* ws = (float*)d_ws;
  size_t o = 0;
  auto alignup = [&]() { o = (o + 7) & ~(size_t)7; };
  auto allocF = [&](size_t nf) { alignup(); float* p = ws + o; o += nf; return p; };
  auto allocH = [&](size_t nh) {
    alignup(); _Float16* p = (_Float16*)(ws + o); o += (nh + 1) / 2; return p;
  };
  float*     g      = allocF((size_t)B * 4 * H);   // gate buffer [B,4H] f32
  float*     ccur   = allocF((size_t)B * H);       // cell state f32
  _Float16*  hcur   = allocH((size_t)B * H);       // hidden state f16
  _Float16*  h0all  = allocH((size_t)B * T * H);   // layer-0 h, f16
  // f16 staged weights (persistent)
  _Float16* Wih0p = allocH((size_t)4 * H * Dp);    // [1024, 768]
  _Float16* Whh0h = allocH((size_t)4 * H * H);
  _Float16* Wih1h = allocH((size_t)4 * H * H);
  _Float16* Whh1h = allocH((size_t)4 * H * H);
  _Float16* Wp[NB][2][5];
  const int wN[5]  = {HID, HID, HID, HID, D};
  const int wK[5]  = {H + D, HID, HID, HID, HID};
  const int wKp[5] = {H + Dp, HIDp, HIDp, HIDp, HIDp};
  for (int blk = 0; blk < NB; ++blk)
    for (int net = 0; net < 2; ++net)
      for (int l = 0; l < 5; ++l)
        Wp[blk][net][l] = allocH((size_t)wN[l] * wKp[l]);
  alignup();
  const size_t arena = o;
  // stage 1 (LSTM) view of arena
  _Float16* xbt   = (_Float16*)(ws + arena);              // [B*T, Dp] f16
  float*    xproj = ws + arena + ((size_t)B * T * Dp + 1) / 2;  // [B*T, 4H] f32
  // stage 2 (log-prob) view of arena
  {} // laid out below with explicit offsets
  size_t o2 = arena;
  auto a2F = [&](size_t nf) { o2 = (o2 + 7) & ~(size_t)7; float* p = ws + o2; o2 += nf; return p; };
  auto a2H = [&](size_t nh) { o2 = (o2 + 7) & ~(size_t)7; _Float16* p = (_Float16*)(ws + o2); o2 += (nh + 1) / 2; return p; };
  float*    ulp   = a2F((size_t)B * Dp);
  _Float16* mxlp  = a2H((size_t)B * Dp);
  float*    slp   = a2F((size_t)B * Dp);
  float*    tlp   = a2F((size_t)B * Dp);
  float*    sumld = a2F((size_t)B * Dp);
  _Float16* hA2   = a2H((size_t)B * HIDp);
  _Float16* hB2   = a2H((size_t)B * HIDp);
  // stage 3 (sampling) view of arena
  size_t o3 = arena;
  auto a3F = [&](size_t nf) { o3 = (o3 + 7) & ~(size_t)7; float* p = ws + o3; o3 += nf; return p; };
  auto a3H = [&](size_t nh) { o3 = (o3 + 7) & ~(size_t)7; _Float16* p = (_Float16*)(ws + o3); o3 += (nh + 1) / 2; return p; };
  float*    us   = a3F((size_t)SB * Dp);
  _Float16* mxs  = a3H((size_t)SB * Dp);
  float*    ss   = a3F((size_t)SB * Dp);
  float*    ts   = a3F((size_t)SB * Dp);
  _Float16* hA3  = a3H((size_t)SB * HIDp);
  _Float16* hB3  = a3H((size_t)SB * HIDp);

  auto gemmGrid = [](int M, int N) {
    long long waves = (long long)((M + 15) / 16) * ((N + 63) / 64);
    return dim3((unsigned)((waves + 7) / 8));
  };
  auto gemm = [&](const _Float16* A0, int lda0, int K0, int rm,
                  const _Float16* A1, int lda1, int Kt,
                  const _Float16* W, int ldw, const float* bias,
                  const float* Cin, int ldcin,
                  void* C, int ldc, int M, int N, int act, int outHalf) {
    k_gemm<<<gemmGrid(M, N), WGT, 0, stream>>>(
        A0, lda0, K0, rm, A1, lda1, Kt, W, ldw, bias, Cin, ldcin,
        C, ldc, M, N, act, outHalf);
  };

  // ---------------- stage weights into zero-padded f16 copies ---------------
  k_padW<<<G1((long long)4 * H * Dp), WGT, 0, stream>>>(Wih0, Wih0p, 4 * H, D, Dp);
  k_padW<<<G1((long long)4 * H * H), WGT, 0, stream>>>(Whh0, Whh0h, 4 * H, H, H);
  k_padW<<<G1((long long)4 * H * H), WGT, 0, stream>>>(Wih1, Wih1h, 4 * H, H, H);
  k_padW<<<G1((long long)4 * H * H), WGT, 0, stream>>>(Whh1, Whh1h, 4 * H, H, H);
  for (int blk = 0; blk < NB; ++blk)
    for (int net = 0; net < 2; ++net)
      for (int l = 0; l < 5; ++l)
        k_padW<<<G1((long long)wN[l] * wKp[l]), WGT, 0, stream>>>(
            fw(blk, net, l), Wp[blk][net][l], wN[l], wK[l], wKp[l]);

  // ============================ LSTM =======================================
  // past[B,D,T] -> xbt[B*T, Dp] f16 (zero-padded)
  k_transpose_past<<<G1((long long)B * T * Dp), WGT, 0, stream>>>(past, xbt, B, D, Dp, T);
  // hoisted input projection, layer 0: xproj = xbt @ Wih0p^T + bih0
  gemm(xbt, Dp, Dp, 0, nullptr, 0, Dp, Wih0p, Dp, bih0, nullptr, 0,
       xproj, 4 * H, B * T, 4 * H, 0, 0);
  // layer-0 recurrence
  k_fillh<<<G1((long long)B * H), WGT, 0, stream>>>(hcur, (long long)B * H);
  k_fill<<<G1((long long)B * H), WGT, 0, stream>>>(ccur, (long long)B * H, 0.0f);
  for (int t = 0; t < T; ++t) {
    gemm(hcur, H, H, 0, nullptr, 0, H, Whh0h, H, bhh0,
         xproj + (size_t)t * 4 * H, T * 4 * H, g, 4 * H, B, 4 * H, 0, 0);
    k_lstm_cell<<<G1((long long)B * H), WGT, 0, stream>>>(
        g, ccur, ccur, hcur, h0all + (size_t)t * H, T * H, B * H, 4 * H, H);
  }
  // hoisted input projection, layer 1: xproj = h0all @ Wih1h^T + bih1
  gemm(h0all, H, H, 0, nullptr, 0, H, Wih1h, H, bih1, nullptr, 0,
       xproj, 4 * H, B * T, 4 * H, 0, 0);
  // layer-1 recurrence (only final h needed -> cond)
  k_fillh<<<G1((long long)B * H), WGT, 0, stream>>>(hcur, (long long)B * H);
  k_fill<<<G1((long long)B * H), WGT, 0, stream>>>(ccur, (long long)B * H, 0.0f);
  for (int t = 0; t < T; ++t) {
    gemm(hcur, H, H, 0, nullptr, 0, H, Whh1h, H, bhh1,
         xproj + (size_t)t * 4 * H, T * 4 * H, g, 4 * H, B, 4 * H, 0, 0);
    k_lstm_cell<<<G1((long long)B * H), WGT, 0, stream>>>(
        g, ccur, ccur, hcur, nullptr, 0, B * H, 4 * H, H);
  }
  const _Float16* cond = hcur;    // [B, H] f16

  // coupling MLP runner: input concat [cond | mx] handled inside the GEMM
  auto runMLP = [&](int R, int rowmod, const _Float16* mx, int blk, int net,
                    int act, float* outbuf, _Float16* hA, _Float16* hB) {
    gemm(cond, H, H, rowmod, mx, Dp, H + Dp, Wp[blk][net][0], H + Dp,
         fb(blk, net, 0), nullptr, 0, hA, HIDp, R, HID, act, 1);
    gemm(hA, HIDp, HIDp, 0, nullptr, 0, HIDp, Wp[blk][net][1], HIDp,
         fb(blk, net, 1), nullptr, 0, hB, HIDp, R, HID, act, 1);
    gemm(hB, HIDp, HIDp, 0, nullptr, 0, HIDp, Wp[blk][net][2], HIDp,
         fb(blk, net, 2), nullptr, 0, hA, HIDp, R, HID, act, 1);
    gemm(hA, HIDp, HIDp, 0, nullptr, 0, HIDp, Wp[blk][net][3], HIDp,
         fb(blk, net, 3), nullptr, 0, hB, HIDp, R, HID, act, 1);
    gemm(hB, HIDp, HIDp, 0, nullptr, 0, HIDp, Wp[blk][net][4], HIDp,
         fb(blk, net, 4), nullptr, 0, outbuf, Dp, R, D, 0, 0);
  };

  // ====================== flow log-prob (loss) =============================
  k_copy_pad<<<G1((long long)B * Dp), WGT, 0, stream>>>(ulp, tgt, B, D, Dp);
  k_fill<<<G1((long long)B * Dp), WGT, 0, stream>>>(sumld, (long long)B * Dp, 0.0f);
  k_fillh<<<G1((long long)B * HIDp), WGT, 0, stream>>>(hA2, (long long)B * HIDp);
  k_fillh<<<G1((long long)B * HIDp), WGT, 0, stream>>>(hB2, (long long)B * HIDp);
  for (int i = 0; i < NB; ++i) {
    const int par = (i % 2 == 0) ? 1 : 0;
    k_maskmul<<<G1((long long)B * Dp), WGT, 0, stream>>>(
        ulp, mxlp, (long long)B * Dp, D, Dp, par);
    runMLP(B, 0, mxlp, i, 0, 1, slp, hA2, hB2);   // s-net, tanh
    runMLP(B, 0, mxlp, i, 1, 2, tlp, hA2, hB2);   // t-net, relu
    k_cpl_fwd<<<G1((long long)B * Dp), WGT, 0, stream>>>(
        ulp, sumld, slp, tlp, (long long)B * Dp, D, Dp, par);
  }
  k_fill<<<1, WGT, 0, stream>>>(out, 2, 0.0f);    // zero loss & mse accumulators
  k_loss_reduce<<<256, WGT, 0, stream>>>(ulp, sumld, mask, lossp,
                                         (long long)B * D, D, Dp, 1.0f / (float)B);

  // ====================== flow sampling (mse, preds) =======================
  k_randn<<<G1((long long)SB * Dp), WGT, 0, stream>>>(us, (long long)SB * Dp, 0x1234567u);
  k_fillh<<<G1((long long)SB * HIDp), WGT, 0, stream>>>(hA3, (long long)SB * HIDp);
  k_fillh<<<G1((long long)SB * HIDp), WGT, 0, stream>>>(hB3, (long long)SB * HIDp);
  for (int i = NB - 1; i >= 0; --i) {
    const int par = (i % 2 == 0) ? 1 : 0;
    k_maskmul<<<G1((long long)SB * Dp), WGT, 0, stream>>>(
        us, mxs, (long long)SB * Dp, D, Dp, par);
    runMLP(SB, B, mxs, i, 0, 1, ss, hA3, hB3);    // cond tiled via rowmod=B
    runMLP(SB, B, mxs, i, 1, 2, ts, hA3, hB3);
    k_cpl_inv<<<G1((long long)SB * Dp), WGT, 0, stream>>>(
        us, ss, ts, (long long)SB * Dp, D, Dp, par);
  }
  k_store_preds<<<G1((long long)SB * D), WGT, 0, stream>>>(
      us, mask, preds, (long long)SB * D, D, Dp, B);
  k_mse_reduce<<<256, WGT, 0, stream>>>(preds, tgt, mask, msep,
                                        B * D, NS, 1.0f / (float)(B * D));
}
